// InnerNetLSTMCell_78769700208974
// MI455X (gfx1250) — matrix operand order repaired and checked by reading, not compiled, not run on hardware
//
#include <hip/hip_runtime.h>
#include <hip/hip_bf16.h>
#include <math.h>

// ---- problem constants (from reference) ----
#define BATCH 4096
#define INSZ  1024
#define HSZ   1024
#define KIN   32
#define KDIM  2048          // IN + H
#define NDIM  5120          // 3H (gates) + 2H (cell)

typedef __bf16 bf16_t;
typedef __attribute__((ext_vector_type(16))) __bf16 v16bf;
typedef __attribute__((ext_vector_type(8)))  __bf16 bf16x8;
typedef __attribute__((ext_vector_type(4)))  __bf16 bf16x4;
typedef __attribute__((ext_vector_type(8)))  float  v8f;
typedef int v4i __attribute__((vector_size(16)));   // matches builtin param type

#define AS1 __attribute__((address_space(1)))
#define AS3 __attribute__((address_space(3)))

#if defined(__has_builtin)
# if __has_builtin(__builtin_amdgcn_global_load_async_to_lds_b128)
#  define HAVE_ASYNC_LDS 1
# endif
# if __has_builtin(__builtin_amdgcn_s_wait_asynccnt)
#  define HAVE_WAIT_ASYNC 1
# endif
#endif

template <int N>
__device__ __forceinline__ void wait_asynccnt() {
#if defined(HAVE_WAIT_ASYNC)
    __builtin_amdgcn_s_wait_asynccnt(N);
#else
    asm volatile("s_wait_asynccnt %0" ::"n"(N));
#endif
}

__device__ __forceinline__ void async_copy16(const bf16_t* g, bf16_t* l) {
#if defined(HAVE_ASYNC_LDS)
    // Builtin signature: (v4i addrspace(1)*, v4i addrspace(3)*, imm offset, imm cpol).
    // AS3 pointers are 32-bit LDS offsets == low 32 bits of the generic pointer.
    __builtin_amdgcn_global_load_async_to_lds_b128(
        (AS1 v4i*)(uintptr_t)g, (AS3 v4i*)(uint32_t)(uintptr_t)l, 0, 0);
#else
    (void)g; (void)l;
#endif
}

// ------------------------------------------------------------------
// Pack [x | h_prev] (fp32) into bf16 row-major [BATCH][KDIM].
// ------------------------------------------------------------------
__global__ void pack_combined(const float* __restrict__ x,
                              const float* __restrict__ h,
                              bf16_t* __restrict__ out) {
    int idx = blockIdx.x * blockDim.x + threadIdx.x;   // [0, BATCH*KDIM/4)
    int row = idx >> 9;                                 // KDIM/4 = 512
    int c4  = idx & 511;
    float4 v;
    if (c4 < 256)  v = ((const float4*)(x + (size_t)row * INSZ))[c4];
    else           v = ((const float4*)(h + (size_t)row * HSZ))[c4 - 256];
    bf16x4 o = { (bf16_t)v.x, (bf16_t)v.y, (bf16_t)v.z, (bf16_t)v.w };
    *(bf16x4*)(out + (size_t)idx * 4) = o;
}

// ------------------------------------------------------------------
// Pack Wg (rows 0..3071) and Wc (rows 3072..5119) into bf16 [NDIM][KDIM].
// ------------------------------------------------------------------
__global__ void pack_weights(const float* __restrict__ Wg,
                             const float* __restrict__ Wc,
                             bf16_t* __restrict__ out) {
    int idx = blockIdx.x * blockDim.x + threadIdx.x;   // [0, NDIM*KDIM/4)
    int row = idx >> 9;
    int c4  = idx & 511;
    const float* src = (row < 3 * HSZ) ? (Wg + (size_t)row * KDIM)
                                       : (Wc + (size_t)(row - 3 * HSZ) * KDIM);
    float4 v = ((const float4*)src)[c4];
    bf16x4 o = { (bf16_t)v.x, (bf16_t)v.y, (bf16_t)v.z, (bf16_t)v.w };
    *(bf16x4*)(out + (size_t)idx * 4) = o;
}

// ------------------------------------------------------------------
// bf16 WMMA GEMM: acts[4096][5120] = combined[4096][2048] * W[5120][2048]^T
// Block tile 128x256, 256 threads = 8 wave32 waves, each wave 64x64
// (4x4 wmma accumulators). Double-buffered LDS with async global->LDS
// copies (ASYNCcnt) when the toolchain exposes them.
// ------------------------------------------------------------------
__global__ __launch_bounds__(256)
void gemm_bf16(const bf16_t* __restrict__ A,
               const bf16_t* __restrict__ W,
               float* __restrict__ C) {
    constexpr int BM = 128, BN = 256, BK = 32;
    constexpr int LDA = 40;                 // padded LDS row stride (bf16 elems)
    constexpr int NSTAGE = KDIM / BK;       // 64
    __shared__ bf16_t As[2][BM * LDA];
    __shared__ bf16_t Bs[2][BN * LDA];

    const int tid  = threadIdx.x;
    const int lane = tid & 31;
    const int wave = tid >> 5;
    const int wm   = wave >> 2;             // 0..1  -> 64-row slab
    const int wn   = wave & 3;              // 0..3  -> 64-col slab
    const int m0   = blockIdx.y * BM;
    const int n0   = blockIdx.x * BN;
    const int r    = lane & 15;
    const int hsel = lane >> 4;

    v8f acc[4][4] = {};

#if !defined(HAVE_ASYNC_LDS)
    bf16x8 stg[6];                          // register-staged fallback
#endif

    // Per-thread fill assignment: 1536 16B chunks/stage, 6 per thread.
    // j=0..1 -> A tile (512 chunks), j=0..3 -> B tile (1024 chunks).
    auto issue = [&](int buf, int k0) {
#if defined(HAVE_ASYNC_LDS)
        #pragma unroll
        for (int j = 0; j < 2; ++j) {
            int chunk = tid + j * 256, row = chunk >> 2, kc = (chunk & 3) * 8;
            async_copy16(A + (size_t)(m0 + row) * KDIM + k0 + kc,
                         &As[buf][row * LDA + kc]);
        }
        #pragma unroll
        for (int j = 0; j < 4; ++j) {
            int chunk = tid + j * 256, row = chunk >> 2, kc = (chunk & 3) * 8;
            async_copy16(W + (size_t)(n0 + row) * KDIM + k0 + kc,
                         &Bs[buf][row * LDA + kc]);
        }
        (void)buf;
#else
        #pragma unroll
        for (int j = 0; j < 2; ++j) {
            int chunk = tid + j * 256, row = chunk >> 2, kc = (chunk & 3) * 8;
            stg[j] = *(const bf16x8*)(A + (size_t)(m0 + row) * KDIM + k0 + kc);
        }
        #pragma unroll
        for (int j = 0; j < 4; ++j) {
            int chunk = tid + j * 256, row = chunk >> 2, kc = (chunk & 3) * 8;
            stg[2 + j] = *(const bf16x8*)(W + (size_t)(n0 + row) * KDIM + k0 + kc);
        }
        (void)buf;
#endif
    };

#if !defined(HAVE_ASYNC_LDS)
    auto commit = [&](int buf) {
        #pragma unroll
        for (int j = 0; j < 2; ++j) {
            int chunk = tid + j * 256, row = chunk >> 2, kc = (chunk & 3) * 8;
            *(bf16x8*)&As[buf][row * LDA + kc] = stg[j];
        }
        #pragma unroll
        for (int j = 0; j < 4; ++j) {
            int chunk = tid + j * 256, row = chunk >> 2, kc = (chunk & 3) * 8;
            *(bf16x8*)&Bs[buf][row * LDA + kc] = stg[2 + j];
        }
    };
#endif

    issue(0, 0);

    #pragma unroll 2
    for (int ks = 0; ks < NSTAGE; ++ks) {
        const int cur = ks & 1;

#if defined(HAVE_ASYNC_LDS)
        if (ks + 1 < NSTAGE) {
            issue(1 - cur, (ks + 1) * BK);   // overlap next stage DMA
            wait_asynccnt<6>();              // stage ks complete; 6 in flight
        } else {
            wait_asynccnt<0>();
        }
#else
        commit(cur);                          // ds_store stage ks
        if (ks + 1 < NSTAGE) issue(1 - cur, (ks + 1) * BK);
#endif
        __syncthreads();

        // A fragments: lane<16 -> M=r, K {0..7,16..23}; lane>=16 -> K {8..15,24..31}
        v16bf afr[4];
        #pragma unroll
        for (int mt = 0; mt < 4; ++mt) {
            const bf16_t* p = &As[cur][(wm * 64 + mt * 16 + r) * LDA];
            bf16x8 lo = *(const bf16x8*)(p + 8 * hsel);
            bf16x8 hi = *(const bf16x8*)(p + 16 + 8 * hsel);
            afr[mt] = __builtin_shufflevector(lo, hi,
                0,1,2,3,4,5,6,7,8,9,10,11,12,13,14,15);
        }
        // B fragments: lane<16 -> N=r, K {0..7,16..23}; lane>=16 -> K {8..15,24..31}
        v16bf bfr[4];
        #pragma unroll
        for (int nt = 0; nt < 4; ++nt) {
            const bf16_t* p = &Bs[cur][(wn * 64 + nt * 16 + r) * LDA];
            bf16x8 lo = *(const bf16x8*)(p + 8 * hsel);
            bf16x8 hi = *(const bf16x8*)(p + 16 + 8 * hsel);
            bfr[nt] = __builtin_shufflevector(lo, hi,
                0,1,2,3,4,5,6,7,8,9,10,11,12,13,14,15);
        }

        #pragma unroll
        for (int mt = 0; mt < 4; ++mt)
            #pragma unroll
            for (int nt = 0; nt < 4; ++nt)
                acc[mt][nt] = __builtin_amdgcn_wmma_f32_16x16x32_bf16(
                    false, afr[mt], false, bfr[nt],
                    (short)0, acc[mt][nt], false, false);

        __syncthreads();
    }

    // C layout: col = lane&15 ; row = vgpr_idx + 8*(lane>>4)
    #pragma unroll
    for (int mt = 0; mt < 4; ++mt) {
        #pragma unroll
        for (int nt = 0; nt < 4; ++nt) {
            size_t rowbase = (size_t)(m0 + wm * 64 + mt * 16 + 8 * hsel);
            size_t col     = (size_t)(n0 + wn * 64 + nt * 16 + r);
            float* o = C + rowbase * NDIM + col;
            #pragma unroll
            for (int v = 0; v < 8; ++v)
                o[(size_t)v * NDIM] = acc[mt][nt][v];
        }
    }
}

// ------------------------------------------------------------------
// Pointwise epilogue: gates + inner 2->32->1 MLP + LSTM cell update.
// ------------------------------------------------------------------
__global__ void lstm_epilogue(const float* __restrict__ acts,
                              const float* __restrict__ c_prev,
                              const float* __restrict__ bg,
                              const float* __restrict__ bc,
                              const float* __restrict__ W1,
                              const float* __restrict__ b1,
                              const float* __restrict__ W2,
                              const float* __restrict__ b2,
                              float* __restrict__ h_next,
                              float* __restrict__ c_next) {
    int idx = blockIdx.x * blockDim.x + threadIdx.x;   // [0, BATCH*HSZ)
    int b  = idx >> 10;
    int hh = idx & 1023;
    const float* row = acts + (size_t)b * NDIM;

    float gi = row[hh]            + bg[hh];
    float gf = row[HSZ + hh]      + bg[HSZ + hh];
    float go = row[2 * HSZ + hh]  + bg[2 * HSZ + hh];
    float i = 1.0f / (1.0f + __expf(-gi));
    float f = 1.0f / (1.0f + __expf(-gf));
    float o = 1.0f / (1.0f + __expf(-go));

    float p0 = row[3 * HSZ + 2 * hh]     + bc[2 * hh];
    float p1 = row[3 * HSZ + 2 * hh + 1] + bc[2 * hh + 1];

    float g = b2[0];
    #pragma unroll
    for (int k = 0; k < KIN; ++k) {
        float hid = fmaf(W1[2 * k], p0, fmaf(W1[2 * k + 1], p1, b1[k]));
        hid = fmaxf(hid, 0.0f);
        g = fmaf(W2[k], hid, g);
    }

    float c = fmaf(f, c_prev[idx], i * g);
    h_next[idx] = o * tanhf(c);
    c_next[idx] = c;
}

// ------------------------------------------------------------------
extern "C" void kernel_launch(void* const* d_in, const int* in_sizes, int n_in,
                              void* d_out, int out_size, void* d_ws, size_t ws_size,
                              hipStream_t stream) {
    const float* x      = (const float*)d_in[0];
    const float* h_prev = (const float*)d_in[1];
    const float* c_prev = (const float*)d_in[2];
    const float* Wg     = (const float*)d_in[3];
    const float* bg     = (const float*)d_in[4];
    const float* Wc     = (const float*)d_in[5];
    const float* bc     = (const float*)d_in[6];
    const float* W1     = (const float*)d_in[7];
    const float* b1     = (const float*)d_in[8];
    const float* W2     = (const float*)d_in[9];
    const float* b2     = (const float*)d_in[10];

    float* out    = (float*)d_out;                 // h_next then c_next
    float* h_next = out;
    float* c_next = out + (size_t)BATCH * HSZ;

    // workspace layout
    char*   ws       = (char*)d_ws;
    bf16_t* combined = (bf16_t*)ws;                                   // 16.8 MB
    bf16_t* Wpack    = (bf16_t*)(ws + (size_t)BATCH * KDIM * 2);      // 21.0 MB
    float*  acts     = (float*)(ws + (size_t)BATCH * KDIM * 2
                                   + (size_t)NDIM * KDIM * 2);        // 83.9 MB

    {   // 1) pack activations to bf16
        int total = BATCH * KDIM / 4;
        pack_combined<<<total / 256, 256, 0, stream>>>(x, h_prev, combined);
    }
    {   // 2) pack [Wg; Wc] to bf16
        int total = NDIM * KDIM / 4;
        pack_weights<<<total / 256, 256, 0, stream>>>(Wg, Wc, Wpack);
    }
    {   // 3) fused GEMM: acts = combined * Wpack^T
        dim3 grid(NDIM / 256, BATCH / 128);        // 20 x 32
        gemm_bf16<<<grid, 256, 0, stream>>>(combined, Wpack, acts);
    }
    {   // 4) pointwise LSTM epilogue
        int total = BATCH * HSZ;
        lstm_epilogue<<<total / 256, 256, 0, stream>>>(
            acts, c_prev, bg, bc, W1, b1, W2, b2, h_next, c_next);
    }
}
